// GAT_68118181314620
// MI455X (gfx1250) — matrix operand order repaired and checked by reading, not compiled, not run on hardware
//
#include <hip/hip_runtime.h>
#include <cstdint>
#include <cstddef>

#define D 128          // channel count (D_IN == D_H == 128)
#define NEG_SLOPE 0.2f

typedef __attribute__((ext_vector_type(16))) _Float16 v16h;
typedef __attribute__((ext_vector_type(8)))  _Float16 v8h;
typedef __attribute__((ext_vector_type(8)))  float    v8f;

// ---------------------------------------------------------------------------
// Float atomic max via signed/unsigned integer ordering trick (correct for
// any mix of signs; init value is -inf).
// ---------------------------------------------------------------------------
__device__ __forceinline__ void atomic_max_float(float* addr, float v) {
    if (v >= 0.0f) {
        atomicMax((int*)addr, __float_as_int(v));
    } else {
        atomicMin((unsigned int*)addr, __float_as_uint(v));
    }
}

// ---------------------------------------------------------------------------
// One-shot converters (run once per launch; tiny).
//   convert_x : xh[i] = (f16)x[i]                       (layer-0 GEMM input)
//   convert_w : Wt[k*128+n] = (f16)W[n*128+k]           (transposed f16 W)
// ---------------------------------------------------------------------------
__global__ void __launch_bounds__(256)
gat_convert_x(const float* __restrict__ x, _Float16* __restrict__ xh, int n) {
    const int i = blockIdx.x * blockDim.x + threadIdx.x;
    if (i < n) xh[i] = (_Float16)x[i];
}

__global__ void __launch_bounds__(256)
gat_convert_w(const float* __restrict__ W, _Float16* __restrict__ Wt) {
    const int i = blockIdx.x * blockDim.x + threadIdx.x;   // i < 128*128
    const int n = i & (D - 1);
    const int k = i >> 7;
    Wt[(size_t)k * D + n] = (_Float16)W[(size_t)n * D + k];
}

// ---------------------------------------------------------------------------
// Kernel 1: H = X @ W^T  using v_wmma_f32_16x16x32_f16, all-f16 operands.
// One wave computes one 16x16 output tile; K-loop of 4 over K=128.
// A fragment (16x32 f16): lane<16 -> M=lane,    K = {0..7, 16..23} (+kt)
//                         lane>=16 -> M=lane-16, K = {8..15, 24..31} (+kt)
//   Xh row-major f16 -> each K-range is 8 contiguous halves = one b128 load.
// B fragment (32x16 f16): lane%16 -> N column, halves -> K = {kb..kb+7, kb+16..kb+23} (+kt).
//   Wt is [k][n] f16 -> 16 halves strided by D = 16 scalar loads.
// C/D (16x16 f32): VGPR r, lane<16 -> (M=r, N=lane); lane>=16 -> (M=8+r, N=lane-16)
// ---------------------------------------------------------------------------
__global__ void __launch_bounds__(128)
gat_linear_wmma(const _Float16* __restrict__ Xh, const _Float16* __restrict__ Wt,
                float* __restrict__ H, int n_rows) {
    const int wid  = blockIdx.x * (blockDim.x >> 5) + (threadIdx.x >> 5);
    const int n_mt = (n_rows + 15) >> 4;
    const int mt   = wid >> 3;     // row tile
    const int nt   = wid & 7;      // col tile (128/16 = 8)
    if (mt >= n_mt) return;        // whole-wave exit: EXEC stays all-ones for WMMA

    const int lane    = threadIdx.x & 31;
    const int m_local = lane & 15;
    const int kb      = (lane < 16) ? 0 : 8;

    int row = mt * 16 + m_local;
    if (row >= n_rows) row = n_rows - 1;           // clamp (N=50000 is 16-aligned anyway)
    const _Float16* __restrict__ xrow = Xh + (size_t)row * D;

    v8f c = {0.f, 0.f, 0.f, 0.f, 0.f, 0.f, 0.f, 0.f};

    #pragma unroll
    for (int kt = 0; kt < D; kt += 32) {
        // ---- A fragment: two 16B-aligned v8h loads ----
        const v8h a0 = *(const v8h*)(xrow + kt + kb);        // K = kb .. kb+7
        const v8h a1 = *(const v8h*)(xrow + kt + kb + 16);   // K = kb+16 .. kb+23
        v16h A;
        #pragma unroll
        for (int j = 0; j < 8; ++j) { A[j] = a0[j]; A[8 + j] = a1[j]; }

        // ---- B fragment: lane -> N column, halves -> K = {kb..kb+7, kb+16..kb+23} (+kt) ----
        const _Float16* bcol = Wt + (size_t)(kt + kb) * D + nt * 16 + m_local;
        v16h B;
        #pragma unroll
        for (int j = 0; j < 8; ++j) { B[j] = bcol[(size_t)j * D]; B[8 + j] = bcol[(size_t)(16 + j) * D]; }

        c = __builtin_amdgcn_wmma_f32_16x16x32_f16(
                /*neg_a=*/false, A, /*neg_b=*/false, B,
                /*c_mod=*/(short)0, c, /*reuse_a=*/false, /*reuse_b=*/false);
    }

    const int r0  = (lane < 16) ? 0 : 8;
    const int col = nt * 16 + (lane & 15);
    float* __restrict__ hout = H + (size_t)(mt * 16 + r0) * D + col;

    // Uniform fast path: full tile in range (always true when n_rows % 16 == 0,
    // which holds for N = 50000). Stores are lane-coalesced (16 lanes x 64B/row).
    if (mt * 16 + 16 <= n_rows) {
        #pragma unroll
        for (int r = 0; r < 8; ++r) hout[(size_t)r * D] = c[r];
    } else {
        #pragma unroll
        for (int r = 0; r < 8; ++r) {
            if (mt * 16 + r0 + r < n_rows) hout[(size_t)r * D] = c[r];
        }
    }
}

// ---------------------------------------------------------------------------
// Kernel 2: per-node attention dots  hs[i] = h[i].a_src, hd[i] = h[i].a_dst
// One wave per node; float4 loads; wave32 butterfly reduction.
// ---------------------------------------------------------------------------
__global__ void __launch_bounds__(256)
gat_node_dots(const float* __restrict__ h,
              const float* __restrict__ a_src, const float* __restrict__ a_dst,
              float* __restrict__ hs, float* __restrict__ hd, int n_nodes) {
    const int node = blockIdx.x * (blockDim.x >> 5) + (threadIdx.x >> 5);
    if (node >= n_nodes) return;
    const int lane = threadIdx.x & 31;

    const float4 v  = ((const float4*)(h + (size_t)node * D))[lane];
    const float4 s1 = ((const float4*)a_src)[lane];
    const float4 s2 = ((const float4*)a_dst)[lane];

    float ds = v.x * s1.x + v.y * s1.y + v.z * s1.z + v.w * s1.w;
    float dd = v.x * s2.x + v.y * s2.y + v.z * s2.z + v.w * s2.w;
    #pragma unroll
    for (int off = 16; off > 0; off >>= 1) {
        ds += __shfl_xor(ds, off, 32);
        dd += __shfl_xor(dd, off, 32);
    }
    if (lane == 0) {
        hs[node] = ds;
        hd[node] = dd;
    }
}

// ---------------------------------------------------------------------------
// Kernel 3: per-layer init. Zero aggregation target (also clears d_out
// poison), set segment-max to -inf and segment-sum to 0.
// ---------------------------------------------------------------------------
__global__ void __launch_bounds__(256)
gat_init(float* __restrict__ target, float* __restrict__ mmax,
         float* __restrict__ ssum, int n_out, int n_nodes) {
    const int i = blockIdx.x * blockDim.x + threadIdx.x;
    if (i < n_out) target[i] = 0.0f;
    if (i < n_nodes) {
        mmax[i] = -__builtin_inff();
        ssum[i] = 0.0f;
    }
}

// ---------------------------------------------------------------------------
// Kernel 4: edge logits + segment max.  Edge i < E comes from g; i >= E is a
// self-loop (src = dst = i - E), matching the reference concatenation.
// ---------------------------------------------------------------------------
__global__ void __launch_bounds__(256)
gat_edge_max(const int* __restrict__ g, const float* __restrict__ hs,
             const float* __restrict__ hd, float* __restrict__ mmax,
             int e_total, int e_orig) {
    const int i = blockIdx.x * blockDim.x + threadIdx.x;
    if (i >= e_total) return;
    const int s = (i < e_orig) ? g[i]          : (i - e_orig);
    const int d = (i < e_orig) ? g[e_orig + i] : (i - e_orig);
    float e = hs[s] + hd[d];
    e = (e > 0.0f) ? e : NEG_SLOPE * e;
    atomic_max_float(&mmax[d], e);
}

// ---------------------------------------------------------------------------
// Kernel 5: w = exp(e - m[dst]); store w; segment-sum via L2 f32 atomics.
// Logit recomputed (cheaper than a 3.4MB round-trip buffer).
// ---------------------------------------------------------------------------
__global__ void __launch_bounds__(256)
gat_edge_exp(const int* __restrict__ g, const float* __restrict__ hs,
             const float* __restrict__ hd, const float* __restrict__ mmax,
             float* __restrict__ wbuf, float* __restrict__ ssum,
             int e_total, int e_orig) {
    const int i = blockIdx.x * blockDim.x + threadIdx.x;
    if (i >= e_total) return;
    const int s = (i < e_orig) ? g[i]          : (i - e_orig);
    const int d = (i < e_orig) ? g[e_orig + i] : (i - e_orig);
    float e = hs[s] + hd[d];
    e = (e > 0.0f) ? e : NEG_SLOPE * e;
    const float w = __expf(e - mmax[d]);
    wbuf[i] = w;
    atomicAdd(&ssum[d], w);
}

// ---------------------------------------------------------------------------
// Kernel 6: weighted scatter aggregation. One wave per edge; float4 gather of
// h[src]; 4 global_atomic_add_f32 per lane into out[dst] (L2-resident).
// ---------------------------------------------------------------------------
__global__ void __launch_bounds__(256)
gat_edge_aggregate(const int* __restrict__ g, const float* __restrict__ h,
                   const float* __restrict__ wbuf, const float* __restrict__ ssum,
                   float* __restrict__ out, int e_total, int e_orig) {
    const int e = blockIdx.x * (blockDim.x >> 5) + (threadIdx.x >> 5);
    if (e >= e_total) return;
    const int lane = threadIdx.x & 31;
    const int s = (e < e_orig) ? g[e]          : (e - e_orig);
    const int d = (e < e_orig) ? g[e_orig + e] : (e - e_orig);

    const float alpha = wbuf[e] / (ssum[d] + 1e-16f);
    const float4 v = ((const float4*)(h + (size_t)s * D))[lane];
    float* o = out + (size_t)d * D + lane * 4;
    atomicAdd(o + 0, v.x * alpha);
    atomicAdd(o + 1, v.y * alpha);
    atomicAdd(o + 2, v.z * alpha);
    atomicAdd(o + 3, v.w * alpha);
}

// ---------------------------------------------------------------------------
// Kernel 7: epilogue  v = relu(acc + b).
//   Layers 0/1: write f16 activation (next layer's GEMM input) only.
//   Layer 2   : write f32 into d_out (in place over acc).
// ---------------------------------------------------------------------------
__global__ void __launch_bounds__(256)
gat_bias_relu(const float* __restrict__ acc, const float* __restrict__ b,
              float* __restrict__ outf, _Float16* __restrict__ outh, int n) {
    const int i = blockIdx.x * blockDim.x + threadIdx.x;
    if (i >= n) return;
    const float v = fmaxf(acc[i] + b[i & (D - 1)], 0.0f);
    if (outh) outh[i] = (_Float16)v;
    else      outf[i] = v;
}

// ---------------------------------------------------------------------------
// Host orchestration. Inputs (setup_inputs order):
//  [0]=x [1]=g [2]=W0 [3]=as0 [4]=ad0 [5]=b0 [6]=W1 ... [13]=b2
// ---------------------------------------------------------------------------
extern "C" void kernel_launch(void* const* d_in, const int* in_sizes, int n_in,
                              void* d_out, int out_size, void* d_ws, size_t ws_size,
                              hipStream_t stream) {
    const float* x = (const float*)d_in[0];
    const int*   g = (const int*)d_in[1];
    const float* Wl[3] = {(const float*)d_in[2], (const float*)d_in[6], (const float*)d_in[10]};
    const float* As[3] = {(const float*)d_in[3], (const float*)d_in[7], (const float*)d_in[11]};
    const float* Ad[3] = {(const float*)d_in[4], (const float*)d_in[8], (const float*)d_in[12]};
    const float* Bb[3] = {(const float*)d_in[5], (const float*)d_in[9], (const float*)d_in[13]};

    const int nN    = in_sizes[0] / D;   // 50000
    const int E     = in_sizes[1] / 2;   // 800000
    const int e_tot = E + nN;            // with self loops

    // Workspace carve-up (256B aligned slices).
    auto align = [](size_t v) { return (v + 255) & ~(size_t)255; };
    char*  ws  = (char*)d_ws;
    size_t off = 0;
    float*    h    = (float*)(ws + off);    off += align((size_t)nN * D * sizeof(float));
    float*    agg  = (float*)(ws + off);    off += align((size_t)nN * D * sizeof(float));
    _Float16* xh   = (_Float16*)(ws + off); off += align((size_t)nN * D * sizeof(_Float16));
    _Float16* Wt0  = (_Float16*)(ws + off); off += align((size_t)D * D * sizeof(_Float16));
    _Float16* Wt1  = (_Float16*)(ws + off); off += align((size_t)D * D * sizeof(_Float16));
    _Float16* Wt2  = (_Float16*)(ws + off); off += align((size_t)D * D * sizeof(_Float16));
    float*    hs   = (float*)(ws + off);    off += align((size_t)nN * sizeof(float));
    float*    hd   = (float*)(ws + off);    off += align((size_t)nN * sizeof(float));
    float*    mmax = (float*)(ws + off);    off += align((size_t)nN * sizeof(float));
    float*    ssum = (float*)(ws + off);    off += align((size_t)nN * sizeof(float));
    float*    wbuf = (float*)(ws + off);    off += align((size_t)e_tot * sizeof(float));
    (void)off; (void)ws_size;
    _Float16* Wt[3] = {Wt0, Wt1, Wt2};

    const int n_mt       = (nN + 15) / 16;
    const int gemm_waves = n_mt * 8;                 // 16x16 tiles
    const int gemm_blk   = (gemm_waves + 3) / 4;     // 4 waves / 128-thread block
    const int dots_blk   = (nN + 7) / 8;             // 8 waves / 256-thread block
    const int init_n     = nN * D;
    const int init_blk   = (init_n + 255) / 256;
    const int edge_blk   = (e_tot + 255) / 256;
    const int aggr_blk   = (e_tot + 7) / 8;          // wave per edge
    const int br_blk     = (nN * D + 255) / 256;
    const int cw_blk     = (D * D + 255) / 256;

    // One-shot f16 staging: layer-0 input + all three transposed weights.
    gat_convert_x<<<init_blk, 256, 0, stream>>>(x, xh, nN * D);
    for (int l = 0; l < 3; ++l)
        gat_convert_w<<<cw_blk, 256, 0, stream>>>(Wl[l], Wt[l]);

    for (int l = 0; l < 3; ++l) {
        float* target = (l == 2) ? (float*)d_out : agg;

        gat_linear_wmma<<<gemm_blk, 128, 0, stream>>>(xh, Wt[l], h, nN);
        gat_node_dots<<<dots_blk, 256, 0, stream>>>(h, As[l], Ad[l], hs, hd, nN);
        gat_init<<<init_blk, 256, 0, stream>>>(target, mmax, ssum, init_n, nN);
        gat_edge_max<<<edge_blk, 256, 0, stream>>>(g, hs, hd, mmax, e_tot, E);
        gat_edge_exp<<<edge_blk, 256, 0, stream>>>(g, hs, hd, mmax, wbuf, ssum, e_tot, E);
        gat_edge_aggregate<<<aggr_blk, 256, 0, stream>>>(g, h, wbuf, ssum, target, e_tot, E);

        if (l == 2) {
            gat_bias_relu<<<br_blk, 256, 0, stream>>>(target, Bb[l], (float*)d_out,
                                                      (_Float16*)nullptr, nN * D);
        } else {
            gat_bias_relu<<<br_blk, 256, 0, stream>>>(target, Bb[l], (float*)nullptr,
                                                      xh, nN * D);
        }
    }
}
